// DendSeqNetSVHN_45930380263776
// MI455X (gfx1250) — matrix operand order repaired and checked by reading, not compiled, not run on hardware
//
#include <hip/hip_runtime.h>

typedef __attribute__((ext_vector_type(16))) _Float16 v16h;
typedef __attribute__((ext_vector_type(8)))  float    v8f;

#define TT 64
#define BB 128
#define HH 200
#define NTILES 13   // 208 = 13*16 padded N tiles

// LDS byte-offset of a __shared__ object (ptrtoint of addrspace(3) pointer)
__device__ __forceinline__ unsigned lds_offset_of(const void* p) {
  return (unsigned)(unsigned long long)(__attribute__((address_space(3))) const void*)p;
}

// Async copy 16 bytes / lane: global -> LDS (ASYNCcnt tracked)
__device__ __forceinline__ void async_copy_b128(unsigned lds_off, const void* gaddr) {
  asm volatile("global_load_async_to_lds_b128 %0, %1, off"
               :: "v"(lds_off), "v"(gaddr) : "memory");
}
// wait until own-wave ASYNCcnt <= 4: previous 4-op batch complete (in-order)
__device__ __forceinline__ void wait_async_le4() {
  asm volatile("s_wait_asynccnt 0x4" ::: "memory");
}
__device__ __forceinline__ void wait_async_0() {
  asm volatile("s_wait_asynccnt 0x0" ::: "memory");
}

// ---------------------------------------------------------------------------
// Stage 1: pack W[c][n][k] (f32) into f16 WMMA B-fragment order.
// B matrix 32x16 (KxN), k order used here (matches A): lanes 0-15 hold K=0..15 (j=k),
// lanes 16-31 hold K=16..31 (j=k-16); each lane's 16 halves are contiguous.
// Fragment index = ((c*13 + nt)*ktiles + kt), 512 halves per fragment.
// ---------------------------------------------------------------------------
__global__ __launch_bounds__(256)
void pack_w(const float* __restrict__ W, _Float16* __restrict__ Bp,
            int ktiles, int KG, int total) {
  int idx = blockIdx.x * blockDim.x + threadIdx.x;
  if (idx >= total) return;
  int j    = idx & 15;
  int lane = (idx >> 4) & 31;
  int frag = idx >> 9;
  int kt   = frag % ktiles;
  int rest = frag / ktiles;
  int nt   = rest % NTILES;
  int c    = rest / NTILES;
  int n = nt * 16 + (lane & 15);
  int k = kt * 32 + ((lane >> 4) << 4) + j;
  float v = (n < HH) ? W[((size_t)c * HH + n) * KG + k] : 0.0f;
  Bp[idx] = (_Float16)v;
}

// ---------------------------------------------------------------------------
// Stage 2: cur[t, branch, b, h] = x_slice[t,b,c,:] . W[c][h,:] + bias[c][h]
// Block = 8 waves = 8 row tiles (16 consecutive (t,b) rows each, same t).
// Double-buffered async B staging: at k-step kt, the block issues
// GLOBAL_LOAD_ASYNC_TO_LDS_B128 for kt+1 into the alternate LDS buffer
// (exactly 4 full-EXEC async ops per thread -> s_wait_asynccnt 4 == previous
// batch done), then consumes buf[kt&1] with 13 ds-fed 16x16x32 f16 WMMAs,
// software-pipelined (fragment nt+1 loaded before WMMA nt issues).
// A fragment (16-bit A 16x32): lane = m + 16*hi,
//   k = kt*32 + 16*hi + j, gathered from x with f32->f16 cvt.
// KG: full K (320/384), WG: cols per feature row (10/12/10),
// COLOFF: column offset in x's last dim (0/10/22), BR0: branch base (0/3/6).
// ---------------------------------------------------------------------------
template <int KG, int WG, int COLOFF, int BR0>
__global__ __launch_bounds__(256)
void gemm_cur(const float* __restrict__ x, const _Float16* __restrict__ Bp,
              const float* __restrict__ bias, float* __restrict__ cur) {
  constexpr int KT = KG / 32;
  constexpr int CHUNKS = NTILES * 64;            // real 16B chunks per batch (832)
  __shared__ __align__(32) _Float16 ldsB[2][NTILES * 512];

  const int c      = blockIdx.y;                 // 0..2
  const int branch = BR0 + c;                    // 0..8
  const int tid  = threadIdx.x;
  const int lane = tid & 31;
  const int wave = tid >> 5;
  const int row0 = (blockIdx.x * 8 + wave) << 4; // 16-row tile base, same t
  const int m  = lane & 15;
  const int hi = lane >> 4;
  const int row = row0 + m;                      // row = t*128 + b
  const float* __restrict__ xrow = x + ((((size_t)row) * 3 + c) << 10); // [32][32]
  const char* __restrict__ bp_c =
      (const char*)(Bp + ((size_t)(c * NTILES * KT) << 9));
  const unsigned lbase[2] = { lds_offset_of(&ldsB[0][0]),
                              lds_offset_of(&ldsB[1][0]) };

  // issue one 4-op-per-thread async batch staging B fragments of k-step kt
  auto issue_batch = [&](int kt, unsigned buf_base) {
#pragma unroll
    for (int i = 0; i < 4; ++i) {
      int chr = tid + i * 256;                   // 0..1023
      int ch  = (chr < CHUNKS) ? chr : (chr - CHUNKS); // pad: rewrite same data
      int nt     = ch >> 6;                      // fragment
      int within = ch & 63;                      // 16B chunk inside fragment
      const void* g = bp_c + (((size_t)nt * KT + kt) << 10) + (within << 4);
      async_copy_b128(buf_base + (unsigned)(ch << 4), g);
    }
  };

  v8f acc[NTILES] = {};

  issue_batch(0, lbase[0]);                      // prologue

  for (int kt = 0; kt < KT; ++kt) {
    // ---- gather + convert the A fragment (overlaps in-flight async DMA) ----
    v16h a;
#pragma unroll
    for (int j = 0; j < 16; ++j) {
      int k   = kt * 32 + (hi << 4) + j;
      int f   = k / WG;                          // constant divisor -> magic mul
      int col = k - f * WG + COLOFF;
      a[j] = (_Float16)xrow[(f << 5) + col];
    }

    // ---- prefetch next k-step's B into the alternate buffer ----
    if (kt + 1 < KT) {
      issue_batch(kt + 1, lbase[(kt + 1) & 1]);
      wait_async_le4();                          // batch kt complete
    } else {
      wait_async_0();
    }
    __syncthreads();                             // all waves' DMA visible

    // ---- 13 WMMAs fed from LDS, software-pipelined fragment loads ----
    const _Float16* lb = &ldsB[kt & 1][0];
    v16h bcur = *(const v16h*)&lb[lane << 4];
#pragma unroll
    for (int nt = 0; nt < NTILES; ++nt) {
      v16h bnext;
      if (nt + 1 < NTILES)
        bnext = *(const v16h*)&lb[((nt + 1) << 9) + (lane << 4)];
      acc[nt] = __builtin_amdgcn_wmma_f32_16x16x32_f16(
          false, a, false, bcur, (short)0, acc[nt], false, false);
      bcur = bnext;
    }
    __syncthreads();                             // buffer reused at kt+2
  }

  // D layout: VGPR r holds M = r + 8*hi, N = nt*16 + (lane&15)
#pragma unroll
  for (int nt = 0; nt < NTILES; ++nt) {
    int n = nt * 16 + (lane & 15);
    if (n < HH) {
      float bval = bias[c * HH + n];
#pragma unroll
      for (int r = 0; r < 8; ++r) {
        int rr = row0 + r + (hi << 3);
        int tt = rr >> 7, bb = rr & 127;
        cur[(((size_t)tt * 9 + branch) * BB + bb) * HH + n] = acc[nt][r] + bval;
      }
    }
  }
}

// ---------------------------------------------------------------------------
// Stage 3: sequential LIF scan; one block per batch element b.
// Thread h (< 200) keeps vh/ih for all 9 branches in registers.
// Readout: 40 threads, each one (k,o) pair doing a 50-length dot over LDS z.
// ---------------------------------------------------------------------------
__global__ __launch_bounds__(256)
void scan_lif(const float* __restrict__ cur, const float* __restrict__ Wout,
              const float* __restrict__ bout, float* __restrict__ volt) {
  const int b = blockIdx.x;
  const int tid = threadIdx.x;
  __shared__ float zs[HH];
  __shared__ float voacc[40];

  float vh[9], ih[9];
#pragma unroll
  for (int br = 0; br < 9; ++br) { vh[br] = 0.f; ih[br] = 0.f; }
  float vo = 0.f, io = 0.f;
  const int k4 = tid / 10, o = tid - k4 * 10;

  for (int t = 0; t < TT; ++t) {
    if (tid < HH) {
      float zsum = 0.f;
      const float* cp = cur + ((size_t)t * 9 * BB + b) * HH + tid;
#pragma unroll
      for (int br = 0; br < 9; ++br) {
        float cval = cp[(size_t)br * BB * HH];
        float vdec = vh[br] + 0.1f * (ih[br] - vh[br]);
        float idec = 0.8f * ih[br];
        float z = (vdec - 1.0f) > 0.f ? 1.f : 0.f;
        vh[br] = (1.f - z) * vdec;
        ih[br] = idec + cval;
        zsum += z;
      }
      zs[tid] = zsum;
    }
    __syncthreads();
    if (tid < 40) {
      float a = bout[k4 * 10 + o];
      const float* w  = Wout + ((size_t)k4 * 10 + o) * 50;
      const float* zp = zs + k4 * 50;
#pragma unroll
      for (int q = 0; q < 50; ++q) a += zp[q] * w[q];
      vo = vo + 0.1f * (io - vo);
      io = 0.8f * io + a;
      voacc[tid] = vo;
    }
    __syncthreads();
    if (tid < 10)
      volt[((size_t)t * BB + b) * 10 + tid] =
          voacc[tid] + voacc[10 + tid] + voacc[20 + tid] + voacc[30 + tid];
    __syncthreads();
  }
}

// ---------------------------------------------------------------------------
extern "C" void kernel_launch(void* const* d_in, const int* in_sizes, int n_in,
                              void* d_out, int out_size, void* d_ws, size_t ws_size,
                              hipStream_t stream) {
  (void)in_sizes; (void)n_in; (void)out_size; (void)ws_size;
  const float* x    = (const float*)d_in[0];
  const float* WL   = (const float*)d_in[1];
  const float* bL   = (const float*)d_in[2];
  const float* WM   = (const float*)d_in[3];
  const float* bM   = (const float*)d_in[4];
  const float* WR   = (const float*)d_in[5];
  const float* bR   = (const float*)d_in[6];
  const float* Wout = (const float*)d_in[7];
  const float* bout = (const float*)d_in[8];
  float* volt = (float*)d_out;

  char* ws = (char*)d_ws;
  float* cur = (float*)ws;                                  // [64,9,128,200] f32
  size_t curBytes = (size_t)TT * 9 * BB * HH * sizeof(float);
  _Float16* BpL = (_Float16*)(ws + curBytes);
  const size_t szL = (size_t)3 * NTILES * 10 * 512;         // halves
  _Float16* BpM = BpL + szL;
  const size_t szM = (size_t)3 * NTILES * 12 * 512;
  _Float16* BpR = BpM + szM;
  const size_t szR = szL;

  pack_w<<<(unsigned)((szL + 255) / 256), 256, 0, stream>>>(WL, BpL, 10, 320, (int)szL);
  pack_w<<<(unsigned)((szM + 255) / 256), 256, 0, stream>>>(WM, BpM, 12, 384, (int)szM);
  pack_w<<<(unsigned)((szR + 255) / 256), 256, 0, stream>>>(WR, BpR, 10, 320, (int)szR);

  dim3 ggrid(64, 3);  // 64 row-blocks (8 waves x 16 rows), 3 channels
  gemm_cur<320, 10, 0,  0><<<ggrid, 256, 0, stream>>>(x, BpL, bL, cur);
  gemm_cur<384, 12, 10, 3><<<ggrid, 256, 0, stream>>>(x, BpM, bM, cur);
  gemm_cur<320, 10, 22, 6><<<ggrid, 256, 0, stream>>>(x, BpR, bR, cur);

  scan_lif<<<BB, 256, 0, stream>>>(cur, Wout, bout, volt);
}
